// TGCN2_4174708211714
// MI455X (gfx1250) — matrix operand order repaired and checked by reading, not compiled, not run on hardware
//
#include <hip/hip_runtime.h>

// ---------------------------------------------------------------------------
// T-GCN on MI455X (gfx1250).
// Dominant cost: 24 GEMMs  C(4096x2176) = A_hat(4096x4096) * XS(4096x2176),
// bf16 inputs, f32 accumulate via v_wmma_f32_16x16x32_bf16.
// A_hat (bf16, 33.5MB) stays resident in the 192MB L2 across all 24 GEMMs.
// Block tile 128x128, 8 waves (4Mx2N), wave tile 32x64 = 8 WMMA / 12 ds_load.
// K-loop kept at unroll 1 so accumulators stay in place (no v_mov rotation).
// ---------------------------------------------------------------------------

typedef __attribute__((ext_vector_type(16))) __bf16 bf16x16;
typedef __attribute__((ext_vector_type(8)))  float  f32x8;

#define NN      4096           // nodes
#define UU      64             // GRU units
#define BB      32             // batch
#define TT      12             // seq len
#define PP      12             // pre len
#define FF      65             // U+1 features
#define COLS    2080           // B*F
#define COLSP   2176           // padded to 17*128
#define GK      4096           // GEMM K
#define LDG     COLSP

__device__ __forceinline__ unsigned short f2bf(float f) {
    unsigned u = __builtin_bit_cast(unsigned, f);
    unsigned r = u + 0x7FFFu + ((u >> 16) & 1u);   // round-to-nearest-even
    return (unsigned short)(r >> 16);
}

// --- A_hat fp32 -> bf16 -----------------------------------------------------
__global__ __launch_bounds__(256) void cvtA_kernel(const float* __restrict__ A,
                                                   unsigned short* __restrict__ Ab) {
    int g = blockIdx.x * 256 + threadIdx.x;
    Ab[g] = f2bf(A[g]);
}

// --- build XS^T (COLSP rows x 4096 nodes) in bf16 ---------------------------
// row c = b*65+f : f==0 -> Xt[b,m], f>=1 -> state[b,m,f-1]; rows >= 2080 zero
__global__ __launch_bounds__(256) void pack_kernel(const float* __restrict__ Xt,
                                                   const float* __restrict__ st,
                                                   unsigned short* __restrict__ Bt) {
    int m = blockIdx.x * 256 + threadIdx.x;
    int c = blockIdx.y;
    float v = 0.0f;
    if (c < COLS) {
        int b = c / FF, f = c % FF;
        v = (f == 0) ? Xt[b * NN + m] : st[(size_t)(b * NN + m) * UU + (f - 1)];
    }
    Bt[(size_t)c * GK + m] = f2bf(v);
}

// --- bf16 WMMA GEMM:  G(4096 x 2176) = A(4096x4096) * BT^T ------------------
__global__ __launch_bounds__(256) void gemm_bf16_kernel(
        const unsigned short* __restrict__ A,
        const unsigned short* __restrict__ BT,
        float* __restrict__ G) {
    // A region: 128 rows x 40 elems; B region: 128 rows x 40 elems; x2 buffers
    __shared__ __align__(16) unsigned short lds[2 * (128 * 40 + 128 * 40)];
    const int tid   = threadIdx.x;
    const int lane  = tid & 31;
    const int wave  = tid >> 5;
    const int waveM = wave & 3;      // 0..3 -> 32-row strip
    const int waveN = wave >> 2;     // 0..1 -> 64-col strip
    const int rowBase = blockIdx.x * 128;
    const int colBase = blockIdx.y * 128;

    // global -> regs staging: A tile 128x32 and B tile 128x32, each
    // 512 x 16B chunks -> 2 chunks per thread per tile.
    const int lrow = tid >> 2;       // 0..63
    const int lseg = tid & 3;        // 0..3 (8 bf16 each)
    const unsigned short* Ar0 = A  + (size_t)(rowBase + lrow)      * GK + lseg * 8;
    const unsigned short* Ar1 = A  + (size_t)(rowBase + lrow + 64) * GK + lseg * 8;
    const unsigned short* Br0 = BT + (size_t)(colBase + lrow)      * GK + lseg * 8;
    const unsigned short* Br1 = BT + (size_t)(colBase + lrow + 64) * GK + lseg * 8;
    const int sA0 = lrow * 40 + lseg * 8;
    const int sA1 = (lrow + 64) * 40 + lseg * 8;
    const int sB0 = 128 * 40 + lrow * 40 + lseg * 8;
    const int sB1 = 128 * 40 + (lrow + 64) * 40 + lseg * 8;
    const int bufStride = 2 * 128 * 40;            // 10240 ushorts per buffer

    // per-lane fragment addresses (ISA 16-bit A / B VGPR layouts)
    const int half = lane >> 4;
    const int lr   = lane & 15;
    int fa[2], fb[4];
#pragma unroll
    for (int mt = 0; mt < 2; ++mt)
        fa[mt] = (waveM * 32 + mt * 16 + lr) * 40 + half * 8;   // +16 for K-high chunk
#pragma unroll
    for (int nt = 0; nt < 4; ++nt)
        fb[nt] = 128 * 40 + (waveN * 64 + nt * 16 + lr) * 40 + half * 8;  // +16 for K-high chunk

    f32x8 acc[2][4];
#pragma unroll
    for (int mt = 0; mt < 2; ++mt)
#pragma unroll
        for (int nt = 0; nt < 4; ++nt) acc[mt][nt] = {};

    uint4 ra0 = *(const uint4*)(Ar0);
    uint4 ra1 = *(const uint4*)(Ar1);
    uint4 rb0 = *(const uint4*)(Br0);
    uint4 rb1 = *(const uint4*)(Br1);
    Ar0 += 32; Ar1 += 32; Br0 += 32; Br1 += 32;
    *(uint4*)(&lds[sA0]) = ra0;
    *(uint4*)(&lds[sA1]) = ra1;
    *(uint4*)(&lds[sB0]) = rb0;
    *(uint4*)(&lds[sB1]) = rb1;
    __syncthreads();

    int cur = 0;
#pragma unroll 1
    for (int kt = 0; kt < GK / 32; ++kt) {
        if (kt + 1 < GK / 32) {                    // prefetch next K tile
            ra0 = *(const uint4*)(Ar0);
            ra1 = *(const uint4*)(Ar1);
            rb0 = *(const uint4*)(Br0);
            rb1 = *(const uint4*)(Br1);
            Ar0 += 32; Ar1 += 32; Br0 += 32; Br1 += 32;
        }
        union Frag { bf16x16 v; uint4 q[2]; } a[2], b[4];
#pragma unroll
        for (int mt = 0; mt < 2; ++mt) {
            a[mt].q[0] = *(const uint4*)(&lds[cur + fa[mt]]);
            a[mt].q[1] = *(const uint4*)(&lds[cur + fa[mt] + 16]);
        }
#pragma unroll
        for (int nt = 0; nt < 4; ++nt) {
            b[nt].q[0] = *(const uint4*)(&lds[cur + fb[nt]]);
            b[nt].q[1] = *(const uint4*)(&lds[cur + fb[nt] + 16]);
        }
#pragma unroll
        for (int mt = 0; mt < 2; ++mt)
#pragma unroll
            for (int nt = 0; nt < 4; ++nt)
                acc[mt][nt] = __builtin_amdgcn_wmma_f32_16x16x32_bf16(
                    false, a[mt].v, false, b[nt].v, (short)0, acc[mt][nt], false, false);

        cur ^= bufStride;
        if (kt + 1 < GK / 32) {
            *(uint4*)(&lds[cur + sA0]) = ra0;
            *(uint4*)(&lds[cur + sA1]) = ra1;
            *(uint4*)(&lds[cur + sB0]) = rb0;
            *(uint4*)(&lds[cur + sB1]) = rb1;
        }
        __syncthreads();
    }

    // C layout: VGPR i -> (M=i, N=lane) for lanes 0..15, (M=i+8, N=lane-16) else
    const int row0 = rowBase + waveM * 32 + (lane >> 4) * 8;
#pragma unroll
    for (int mt = 0; mt < 2; ++mt)
#pragma unroll
        for (int nt = 0; nt < 4; ++nt) {
            const int col = colBase + waveN * 64 + nt * 16 + (lane & 15);
#pragma unroll
            for (int i = 0; i < 8; ++i)
                G[(size_t)(row0 + mt * 16 + i) * LDG + col] = acc[mt][nt][i];
        }
}

// --- gates: r,u = sigmoid(G1 @ W1 + b1) with T-GCN's flat-split quirk -------
// flat i in [0, N*U): r from node i/128 ch i%128; u from node 2048+i/128.
// Writes u, and writes r*state straight into the bf16 B^T pack (rows f>=1).
__global__ __launch_bounds__(256) void gates_kernel(
        const float* __restrict__ G, const float* __restrict__ W1,
        const float* __restrict__ b1, const float* __restrict__ st,
        float* __restrict__ ubuf, unsigned short* __restrict__ Bt) {
    __shared__ float sW[FF * 2 * UU];
    for (int i = threadIdx.x; i < FF * 2 * UU; i += 256) sW[i] = W1[i];
    __syncthreads();
    int g = blockIdx.x * 256 + threadIdx.x;        // b*262144 + i
    int b = g >> 18;
    int i = g & (NN * UU - 1);
    int nr = i >> 7, j = i & 127;
    const float* gr = G + (size_t)nr * LDG + b * FF;
    const float* gu = G + (size_t)(2048 + nr) * LDG + b * FF;
    float ar = b1[j], au = b1[j];
#pragma unroll 5
    for (int f = 0; f < FF; ++f) {
        float w = sW[f * 128 + j];
        ar += gr[f] * w;
        au += gu[f] * w;
    }
    float r = 1.0f / (1.0f + __expf(-ar));
    float u = 1.0f / (1.0f + __expf(-au));
    ubuf[g] = u;
    float rs = r * st[g];
    int n = i >> 6, c = i & 63;                    // (node, unit)
    Bt[(size_t)(b * FF + c + 1) * GK + n] = f2bf(rs);
}

// --- update: c = tanh(G2 @ W2 + b2); new = u*st + (1-u)*c -------------------
__global__ __launch_bounds__(256) void update_kernel(
        const float* __restrict__ G, const float* __restrict__ W2,
        const float* __restrict__ b2, const float* __restrict__ st,
        const float* __restrict__ ubuf, float* __restrict__ stNew) {
    __shared__ float sW[FF * UU];
    for (int i = threadIdx.x; i < FF * UU; i += 256) sW[i] = W2[i];
    __syncthreads();
    int g = blockIdx.x * 256 + threadIdx.x;
    int b = g >> 18;
    int i = g & (NN * UU - 1);
    int n = i >> 6, c = i & 63;
    const float* gp = G + (size_t)n * LDG + b * FF;
    float acc = b2[c];
#pragma unroll 5
    for (int f = 0; f < FF; ++f) acc += gp[f] * sW[f * UU + c];
    float cand = tanhf(acc);
    float u = ubuf[g];
    stNew[g] = u * st[g] + (1.0f - u) * cand;
}

// --- output head: out[b,p,n] = sum_u st[b,n,u]*Wo[u,p] + bo[p] --------------
__global__ __launch_bounds__(256) void out_kernel(
        const float* __restrict__ st, const float* __restrict__ Wo,
        const float* __restrict__ bo, float* __restrict__ out) {
    __shared__ float sWo[UU * PP];
    for (int i = threadIdx.x; i < UU * PP; i += 256) sWo[i] = Wo[i];
    __syncthreads();
    int g = blockIdx.x * 256 + threadIdx.x;        // b*4096 + n
    int b = g >> 12, n = g & (NN - 1);
    const float* s = st + (size_t)b * NN * UU + (size_t)n * UU;
    float acc[PP];
#pragma unroll
    for (int p = 0; p < PP; ++p) acc[p] = bo[p];
    for (int u = 0; u < UU; ++u) {
        float sv = s[u];
#pragma unroll
        for (int p = 0; p < PP; ++p) acc[p] += sv * sWo[u * PP + p];
    }
#pragma unroll
    for (int p = 0; p < PP; ++p)
        out[(size_t)(b * PP + p) * NN + n] = acc[p];
}

extern "C" void kernel_launch(void* const* d_in, const int* in_sizes, int n_in,
                              void* d_out, int out_size, void* d_ws, size_t ws_size,
                              hipStream_t stream) {
    (void)in_sizes; (void)n_in; (void)out_size; (void)ws_size;
    const float* A_hat = (const float*)d_in[0];
    const float* X     = (const float*)d_in[1];
    const float* st0   = (const float*)d_in[2];
    const float* W1    = (const float*)d_in[3];
    const float* b1    = (const float*)d_in[4];
    const float* W2    = (const float*)d_in[5];
    const float* b2    = (const float*)d_in[6];
    const float* Wo    = (const float*)d_in[7];
    const float* bo    = (const float*)d_in[8];
    float* out = (float*)d_out;

    // workspace layout (~188 MB total)
    char* ws = (char*)d_ws;
    unsigned short* Ab = (unsigned short*)ws; ws += (size_t)NN * NN * 2;       // 33.5 MB
    unsigned short* Bt = (unsigned short*)ws; ws += (size_t)COLSP * GK * 2;    // 17.8 MB
    float* G    = (float*)ws; ws += (size_t)NN * COLSP * 4;                    // 35.7 MB
    float* stA  = (float*)ws; ws += (size_t)BB * NN * UU * 4;                  // 33.5 MB
    float* stB  = (float*)ws; ws += (size_t)BB * NN * UU * 4;                  // 33.5 MB
    float* ubuf = (float*)ws; ws += (size_t)BB * NN * UU * 4;                  // 33.5 MB

    cvtA_kernel<<<(NN * NN) / 256, 256, 0, stream>>>(A_hat, Ab);
    hipMemcpyAsync(stA, st0, (size_t)BB * NN * UU * 4, hipMemcpyDeviceToDevice, stream);

    float* cur = stA;
    float* nxt = stB;
    const dim3 gemmGrid(NN / 128, COLSP / 128);    // 32 x 17
    for (int t = 0; t < TT; ++t) {
        const float* Xt = X + (size_t)t * BB * NN;
        pack_kernel<<<dim3(NN / 256, COLSP), 256, 0, stream>>>(Xt, cur, Bt);
        gemm_bf16_kernel<<<gemmGrid, 256, 0, stream>>>(Ab, Bt, G);
        gates_kernel<<<(BB * NN * UU) / 256, 256, 0, stream>>>(G, W1, b1, cur, ubuf, Bt);
        gemm_bf16_kernel<<<gemmGrid, 256, 0, stream>>>(Ab, Bt, G);
        update_kernel<<<(BB * NN * UU) / 256, 256, 0, stream>>>(G, W2, b2, cur, ubuf, nxt);
        float* tmp = cur; cur = nxt; nxt = tmp;
    }
    out_kernel<<<(BB * NN) / 256, 256, 0, stream>>>(cur, Wo, bo, out);
}
